// ClusterNet_58969900974217
// MI455X (gfx1250) — compile-verified, matrix operand order checked by reading, not run on hardware
//
#include <hip/hip_runtime.h>
#include <cstdint>
#include <cstddef>

// ---------------------------------------------------------------------------
// Soft k-means (ClusterNet) for MI455X / gfx1250.
//   x:[n,512] f32, init_mu:[128,512] f32, schedule = 3 EM iterations + final
//   assignment (reference: _cluster(x,init,2) then _cluster(x,mu,1)).
// Strategy: normalize x once -> bf16 (102MB, L2-resident), all GEMMs via
// v_wmma_f32_16x16x32_bf16 with f32 accumulation; softmax/sums/divide in f32.
// mu operand staged into LDS once per workgroup via the Tensor Data Mover
// (tensor_load_to_lds, TENSORcnt), LDS region aliased with the x^T tile.
// ---------------------------------------------------------------------------

typedef __attribute__((ext_vector_type(16))) __bf16 v16bf;
typedef __attribute__((ext_vector_type(8)))  __bf16 v8bf;
typedef __attribute__((ext_vector_type(8)))  float  v8f;
typedef __attribute__((ext_vector_type(4)))  unsigned int u32x4;
typedef __attribute__((ext_vector_type(8)))  int  i32x8;
typedef __attribute__((ext_vector_type(4)))  int  i32x4;

#define BETA  5.0f
#define DDIM  512
#define KCL   128
#define TILE  128
#define LDSB  ((KCL * TILE + DDIM * TILE) * 2)   // rT (32KB) + mu/xT alias (128KB)

__device__ __forceinline__ v16bf load_frag(const __bf16* p0, const __bf16* p1) {
  v16bf r;
  ((v8bf*)&r)[0] = *(const v8bf*)p0;
  ((v8bf*)&r)[1] = *(const v8bf*)p1;
  return r;
}

// ---- row-wise L2 normalize, f32 -> bf16 -----------------------------------
__global__ void __launch_bounds__(128) normalize_k(const float* __restrict__ x,
                                                   __bf16* __restrict__ xn, int n) {
  const int row = blockIdx.x;
  if (row >= n) return;
  const float4 v = *(const float4*)(x + (size_t)row * DDIM + threadIdx.x * 4);
  float ss = v.x * v.x + v.y * v.y + v.z * v.z + v.w * v.w;
  #pragma unroll
  for (int off = 16; off > 0; off >>= 1) ss += __shfl_xor(ss, off, 32);
  __shared__ float ws[4];
  if ((threadIdx.x & 31) == 0) ws[threadIdx.x >> 5] = ss;
  __syncthreads();
  const float rn = rsqrtf(ws[0] + ws[1] + ws[2] + ws[3]);
  __bf16* o = xn + (size_t)row * DDIM + threadIdx.x * 4;
  o[0] = (__bf16)(v.x * rn); o[1] = (__bf16)(v.y * rn);
  o[2] = (__bf16)(v.z * rn); o[3] = (__bf16)(v.w * rn);
}

// ---- f32 -> bf16 copy (initial mu) ----------------------------------------
__global__ void __launch_bounds__(256) cvt_bf16_k(const float* __restrict__ src,
                                                  __bf16* __restrict__ dst, int count) {
  const int i = blockIdx.x * 256 + threadIdx.x;
  if (i < count) dst[i] = (__bf16)src[i];
}

// ---- zero accumulators -----------------------------------------------------
__global__ void __launch_bounds__(256) zero_k(float* __restrict__ acc,
                                              float* __restrict__ cr, int na, int nc) {
  const int i = blockIdx.x * 256 + threadIdx.x;
  if (i < na) acc[i] = 0.0f;
  if (i < nc) cr[i]  = 0.0f;
}

// ---- mu update: mu = acc / cluster_r, requantize bf16, optional f32 out ----
__global__ void __launch_bounds__(256) mu_update_k(const float* __restrict__ acc,
                                                   const float* __restrict__ cr,
                                                   __bf16* __restrict__ mu_bf,
                                                   float* __restrict__ mu_out,
                                                   int write_f32) {
  const int i = blockIdx.x * 256 + threadIdx.x;
  if (i >= KCL * DDIM) return;
  const float v = acc[i] / cr[i >> 9];     // i>>9 == i / DDIM
  mu_bf[i] = (__bf16)v;
  if (write_f32) mu_out[i] = v;
}

// ---- TDM: DMA mu[128][512] bf16 from global into LDS (2D descriptor) -------
__device__ __forceinline__ void tdm_load_mu(const __bf16* mu_global, __bf16* mu_lds) {
  const unsigned long long ga = (unsigned long long)(uintptr_t)mu_global;
  const unsigned int ldsoff = (unsigned int)(uintptr_t)mu_lds;  // flat low 32 = LDS offset

  u32x4 g0;
  g0[0] = 1u;                                   // count=1, user descriptor
  g0[1] = ldsoff;                               // lds_addr (bytes)
  g0[2] = (unsigned int)ga;                     // global_addr[31:0]
  g0[3] = (unsigned int)((ga >> 32) & 0x1FFFFFFu) | (2u << 30);  // addr[56:32] | type=2

  i32x8 g1;
  g1[0] = 0x10000;                // workgroup_mask=0, data_size=1 (2 bytes)
  g1[1] = (int)(512u << 16);      // tensor_dim0 = 512 (bits 79:48, low half)
  g1[2] = (int)(128u << 16);      // tensor_dim0 hi=0 | tensor_dim1 = 128 (low half)
  g1[3] = (int)(512u << 16);      // tensor_dim1 hi=0 | tile_dim0 = 512
  g1[4] = 128;                    // tile_dim1 = 128, tile_dim2 = 0
  g1[5] = 512;                    // tensor_dim0_stride = 512 (low 32)
  g1[6] = 0;                      // stride0 hi | tensor_dim1_stride lo
  g1[7] = 0;

  i32x4 z4 = {};                  // 2D tensor: groups 2/3 unused
  i32x8 z8 = {};
  __builtin_amdgcn_tensor_load_to_lds(g0, g1, z4, z4, z8, 0);
  __builtin_amdgcn_s_wait_tensorcnt(0);
}

// ---- fused: logits^T (WMMA) -> softmax -> rT -> cluster_r -> r^T x (WMMA) --
__global__ void __launch_bounds__(256) fused_iter(
    const __bf16* __restrict__ xn, const __bf16* __restrict__ mu,
    float* __restrict__ acc, float* __restrict__ cr,
    float* __restrict__ r_out, int n, int do_update, int do_write_r) {
  extern __shared__ __bf16 lds[];
  __bf16* rT  = lds;                     // [128 clusters][128 rows] bf16
  __bf16* muL = lds + KCL * TILE;        // phase 1: mu[128][512] bf16 (TDM-staged)
  __bf16* xT  = muL;                     // phase 2b/3: [512 feats][128 rows] (alias)

  const int w    = threadIdx.x >> 5;     // wave 0..7
  const int lane = threadIdx.x & 31;
  const int lrow = lane & 15;
  const int lhi  = lane >> 4;            // half-wave select (K split)
  const int rowbase = blockIdx.x * TILE;

  // ---------------- Phase 0: TDM-stage mu into LDS (wave 0 only) -----------
  if (w == 0) tdm_load_mu(mu, muL);
  __syncthreads();

  // ---------------- Phase 1: logits^T = mu(128x512) * xn_tile^T ------------
  // Wave w owns rows [rowbase + 16w, +16): 8 M-tiles (clusters) x 1 N-tile.
  const int myrow = rowbase + w * 16 + lrow;
  const int ldrow = (myrow < n) ? myrow : (n - 1);
  const __bf16* xrow = xn + (size_t)ldrow * DDIM;

  v8f zero8 = {};
  v8f cfr[8];
  #pragma unroll
  for (int mt = 0; mt < 8; ++mt) cfr[mt] = zero8;

  for (int ks = 0; ks < DDIM / 32; ++ks) {
    const int f0 = ks * 32 + lhi * 8;
    const v16bf bfrag = load_frag(xrow + f0, xrow + f0 + 16);   // B: x rows as cols
    #pragma unroll
    for (int mt = 0; mt < 8; ++mt) {
      const __bf16* murow = muL + (mt * 16 + lrow) * DDIM;      // A: mu rows (LDS)
      const v16bf afrag = load_frag(murow + f0, murow + f0 + 16);
      cfr[mt] = __builtin_amdgcn_wmma_f32_16x16x32_bf16(
          false, afrag, false, bfrag, (short)0, cfr[mt], false, false);
    }
  }

  // ---------------- Phase 1b: softmax over 128 clusters per row ------------
  // Lane holds 64 clusters of its row; other 64 live in lane^16.
  float mx = -3.0e38f;
  #pragma unroll
  for (int mt = 0; mt < 8; ++mt)
    #pragma unroll
    for (int v = 0; v < 8; ++v) {
      const float t = BETA * cfr[mt][v];
      cfr[mt][v] = t;
      mx = fmaxf(mx, t);
    }
  mx = fmaxf(mx, __shfl_xor(mx, 16, 32));
  float s = 0.0f;
  #pragma unroll
  for (int mt = 0; mt < 8; ++mt)
    #pragma unroll
    for (int v = 0; v < 8; ++v) {
      const float e = __expf(cfr[mt][v] - mx);
      cfr[mt][v] = e;
      s += e;
    }
  s += __shfl_xor(s, 16, 32);
  const float sc = (myrow < n) ? (1.0f / s) : 0.0f;  // tail rows -> r = 0

  // ---------------- Phase 1c: stage rT (bf16), optional f32 r output -------
  const int tr = w * 16 + lrow;
  #pragma unroll
  for (int mt = 0; mt < 8; ++mt)
    #pragma unroll
    for (int v = 0; v < 8; ++v) {
      const float r = cfr[mt][v] * sc;
      const int c = mt * 16 + v + 8 * lhi;
      rT[c * TILE + tr] = (__bf16)r;
      if (do_write_r && myrow < n) r_out[(size_t)myrow * KCL + c] = r;
    }

  if (!do_update) return;   // uniform across block
  __syncthreads();          // rT published; muL dead -> region becomes xT

  // ---------------- Phase 2: cluster_r partial sums ------------------------
  if (threadIdx.x < KCL) {
    float ssum = 0.0f;
    #pragma unroll 8
    for (int j = 0; j < TILE; ++j) ssum += (float)rT[threadIdx.x * TILE + j];
    atomicAdd(&cr[threadIdx.x], ssum);
  }

  // ---------------- Phase 2b: stage x tile transposed (xT[feat][row]) ------
  for (int u = threadIdx.x; u < TILE * (DDIM / 8); u += 256) {
    const int row = u / (DDIM / 8);
    const int fc  = (u % (DDIM / 8)) * 8;
    int gr = rowbase + row;
    if (gr >= n) gr = n - 1;               // rT is zero there, so value is moot
    const v8bf xv = *(const v8bf*)(xn + (size_t)gr * DDIM + fc);
    #pragma unroll
    for (int j = 0; j < 8; ++j) xT[(fc + j) * TILE + row] = xv[j];
  }
  __syncthreads();

  // ---------------- Phase 3: acc[c,f] += sum_rows r[row,c]*x[row,f] --------
  // Wave w covers clusters [64*(w&1), +64) x feats [128*(w>>1), +128).
  const int cb = (w & 1) * 64;
  const int fb = (w >> 1) * 128;
  for (int nt = 0; nt < 8; ++nt) {
    const int f0 = fb + nt * 16;
    v8f cf2[4];
    #pragma unroll
    for (int mt = 0; mt < 4; ++mt) cf2[mt] = zero8;
    #pragma unroll
    for (int kk = 0; kk < 4; ++kk) {
      const int kb = kk * 32 + lhi * 8;
      const __bf16* bp = xT + (f0 + lrow) * TILE + kb;   // B: x^T cols = feats
      const v16bf bfrag = load_frag(bp, bp + 16);
      #pragma unroll
      for (int mt = 0; mt < 4; ++mt) {
        const __bf16* ap = rT + (cb + mt * 16 + lrow) * TILE + kb;  // A: r^T
        const v16bf afrag = load_frag(ap, ap + 16);
        cf2[mt] = __builtin_amdgcn_wmma_f32_16x16x32_bf16(
            false, afrag, false, bfrag, (short)0, cf2[mt], false, false);
      }
    }
    #pragma unroll
    for (int mt = 0; mt < 4; ++mt)
      #pragma unroll
      for (int v = 0; v < 8; ++v) {
        const int c = cb + mt * 16 + v + 8 * lhi;
        atomicAdd(&acc[(size_t)c * DDIM + f0 + lrow], cf2[mt][v]);
      }
  }
}

// ---------------------------------------------------------------------------
extern "C" void kernel_launch(void* const* d_in, const int* in_sizes, int n_in,
                              void* d_out, int out_size, void* d_ws, size_t ws_size,
                              hipStream_t stream) {
  const float* x       = (const float*)d_in[0];
  const float* init_mu = (const float*)d_in[1];
  // d_in[2] (num_iter) lives on device; schedule is fixed by the reference:
  // _cluster(.,2) then _cluster(.,1) -> 3 update iterations + final assign.
  const int n = in_sizes[0] / DDIM;

  char*  ws  = (char*)d_ws;
  size_t off = 0;
  __bf16* xn = (__bf16*)(ws + off);
  off += (size_t)n * DDIM * sizeof(__bf16);       off = (off + 255) & ~(size_t)255;
  __bf16* mu_bf = (__bf16*)(ws + off);
  off += (size_t)KCL * DDIM * sizeof(__bf16);     off = (off + 255) & ~(size_t)255;
  float* acc = (float*)(ws + off);
  off += (size_t)KCL * DDIM * sizeof(float);      off = (off + 255) & ~(size_t)255;
  float* cr = (float*)(ws + off);

  float* mu_out = (float*)d_out;
  float* r_out  = (float*)d_out + (size_t)KCL * DDIM;

  (void)hipFuncSetAttribute((const void*)fused_iter,
                            hipFuncAttributeMaxDynamicSharedMemorySize, LDSB);

  const int kd_blocks  = (KCL * DDIM + 255) / 256;
  const int row_blocks = (n + TILE - 1) / TILE;

  normalize_k<<<n, 128, 0, stream>>>(x, xn, n);
  cvt_bf16_k<<<kd_blocks, 256, 0, stream>>>(init_mu, mu_bf, KCL * DDIM);

  for (int it = 0; it < 3; ++it) {
    zero_k<<<kd_blocks, 256, 0, stream>>>(acc, cr, KCL * DDIM, KCL);
    fused_iter<<<row_blocks, 256, LDSB, stream>>>(xn, mu_bf, acc, cr, nullptr, n,
                                                  /*do_update=*/1, /*do_write_r=*/0);
    mu_update_k<<<kd_blocks, 256, 0, stream>>>(acc, cr, mu_bf, mu_out,
                                               /*write_f32=*/(it == 2) ? 1 : 0);
  }
  // final assignment only: r = softmax(beta * x mu^T) written f32 to d_out
  fused_iter<<<row_blocks, 256, LDSB, stream>>>(xn, mu_bf, acc, cr, r_out, n,
                                                /*do_update=*/0, /*do_write_r=*/1);
}
